// FlashFFN_23699629539600
// MI455X (gfx1250) — compile-verified, matrix operand order checked by reading, not run on hardware
//
#include <hip/hip_runtime.h>

// ---------------------------------------------------------------------------
// FlashFFN for MI455X (gfx1250): y = gelu(x@W1 + b1) @ W2 -> nan_to_num -> +b2
//   T = B*S = 4096, D_MODEL = 4096, D_FF = 16384, all f32 in/out.
// HBM-bound (~800MB effective traffic @ 23.3 TB/s ~= 35us; bf16-WMMA compute
// is <1us), so the design optimizes data movement:
//   * bf16 WMMA (v_wmma_f32_16x16x32_bf16) with f32 accumulation
//   * bf16 intermediate h in d_ws (halves the h traffic vs f32)
//   * M-major grid: x (67MB) stays resident in the 192MB L2 while each W
//     column panel is reused by all 32 M-tiles -> weights stream ~once
//   * double-buffered LDS + register prefetch: next K-tile's global loads
//     are in flight during the WMMAs; one barrier per K-step
//   * W staged as 2(K)x4(N) patches: K-pairs pack into dword DS stores
//   * global_prefetch_b8 primes GL2 two K-steps ahead on the weight panel
// ---------------------------------------------------------------------------

typedef __attribute__((ext_vector_type(16))) __bf16        v16bf;
typedef __attribute__((ext_vector_type(8)))  float         v8f;
typedef __attribute__((ext_vector_type(4)))  unsigned int  u32x4;
typedef __attribute__((ext_vector_type(2)))  unsigned int  u32x2;

struct FragBytes { u32x4 lo, hi; };
static_assert(sizeof(v16bf) == 32, "v16bf must be 32 bytes");
static_assert(sizeof(FragBytes) == 32, "FragBytes must be 32 bytes");

#define BM 128
#define BN 128
#define BK 32   // matches WMMA K for bf16

#define T_TOK   4096   // B*S
#define DMODEL  4096
#define DFF     16384

// f32 -> bf16, round-to-nearest-even (manual fallback)
static __device__ inline unsigned short f2bf(float f) {
    unsigned int u = __float_as_uint(f);
    u += 0x7FFFu + ((u >> 16) & 1u);
    return (unsigned short)(u >> 16);
}

// Dependent-context bit_cast so a non-4-byte builtin return type can't break
// the build (discarded if-constexpr branch is not instantiated in a template).
template <typename T>
static __device__ inline unsigned int as_u32_or(T r, unsigned int fallback) {
    if constexpr (sizeof(T) == 4) {
        return __builtin_bit_cast(unsigned int, r);
    } else {
        return fallback;
    }
}

// Pack two f32 into 2xbf16 dword: hardware packed convert if available.
static __device__ inline unsigned int pack2bf(float lo, float hi) {
#if defined(__has_builtin)
#if __has_builtin(__builtin_amdgcn_cvt_pk_bf16_f32)
    return as_u32_or(__builtin_amdgcn_cvt_pk_bf16_f32(lo, hi),
                     (unsigned int)f2bf(lo) | ((unsigned int)f2bf(hi) << 16));
#else
    return (unsigned int)f2bf(lo) | ((unsigned int)f2bf(hi) << 16);
#endif
#else
    return (unsigned int)f2bf(lo) | ((unsigned int)f2bf(hi) << 16);
#endif
}

// jax.nn.gelu (approximate=True), sigmoid form (branch-free):
//   0.5*x*(1+tanh(z)) == x / (1 + exp(-2z)),  z = sqrt(2/pi)*(x + 0.044715 x^3)
static __device__ inline float gelu_tanh(float x) {
    const float c0 = 0.7978845608028654f;  // sqrt(2/pi)
    const float c1 = 0.044715f;
    float z = c0 * (x + c1 * x * x * x);
    return x * __builtin_amdgcn_rcpf(1.0f + __expf(-2.0f * z));
}

// Build one 16x32 bf16 WMMA operand fragment from an LDS tile stored as
// [16 rows][32 bf16] row-major with row stride BK(=32) bf16.
// ISA 16-bit A layout (wave32): lane L holds row M=L%16; lane group g=L/16
// selects K chunks {g*8..g*8+7} (VGPR0-3) and {16+g*8..16+g*8+7} (VGPR4-7).
static __device__ inline v16bf load_frag(const unsigned short* lds,
                                         int row_base, int lane) {
    const int r = row_base + (lane & 15);
    const int g = (lane >> 4) & 1;
    const char* base = (const char*)(lds + r * BK);
    FragBytes fb;
    fb.lo = *(const u32x4*)(base + g * 16);        // K = g*8    .. g*8+7
    fb.hi = *(const u32x4*)(base + 32 + g * 16);   // K = 16+g*8 .. +7
    return __builtin_bit_cast(v16bf, fb);
}

// STAGE 1: A = x (f32), out = h (bf16) = gelu(A@W + b1)
// STAGE 2: A = h (bf16), out = y (f32) = nan_guard(A@W) + b2
template <int STAGE>
__global__ __launch_bounds__(256)
void ffn_gemm_kernel(const void* __restrict__ Araw,
                     const float* __restrict__ W,
                     const float* __restrict__ bias,
                     void* __restrict__ outp,
                     int M, int N, int K) {
    __shared__ __align__(16) unsigned short ldsA[2][BM * BK];  // [row][k]
    __shared__ __align__(16) unsigned short ldsB[2][BN * BK];  // [col][k] (W^T)

    const int tid  = threadIdx.x;
    const int lane = tid & 31;
    const int wave = tid >> 5;
    const int wm   = wave & 3;   // 4 waves tile M: 32 rows each
    const int wn   = wave >> 2;  // 2 waves tile N: 64 cols each

    const int row0 = blockIdx.x * BM;   // M-major grid: W panel L2 reuse
    const int col0 = blockIdx.y * BN;

    v8f acc[2][4] = {};  // 32x64 of C per wave as 8 16x16 f32 fragments

    // ---- register prefetch state for the software pipeline ----
    float4 pa[4];   // STAGE 1: A tile f32
    u32x4  pa2[2];  // STAGE 2: A tile bf16
    float4 pb[4];   // W tile f32: 2 patches of 2(K) x 4(N)

    auto global_load = [&](int k0) {
        if (STAGE == 1) {
            const float* A = (const float*)Araw;
            #pragma unroll
            for (int i = 0; i < 4; ++i) {
                int f4 = i * 256 + tid;       // 1024 float4 in 128x32 tile
                int r  = f4 >> 3;             // 8 float4 per 32-wide row
                int c  = (f4 & 7) * 4;
                pa[i] = *(const float4*)(A + (size_t)(row0 + r) * K + k0 + c);
            }
        } else {
            const unsigned short* A = (const unsigned short*)Araw;  // bf16 h
            #pragma unroll
            for (int i = 0; i < 2; ++i) {
                int ch = i * 256 + tid;       // 512 chunks of 8 bf16
                int r  = ch >> 2;             // 4 chunks per 32-wide row
                int c  = (ch & 3) * 8;
                pa2[i] = *(const u32x4*)(A + (size_t)(row0 + r) * K + k0 + c);
            }
        }
        // W tile 32x128: 512 patches of 2 rows x 4 cols, 2 patches per thread
        #pragma unroll
        for (int i = 0; i < 2; ++i) {
            int p  = i * 256 + tid;
            int pr = p >> 5;                  // row pair 0..15
            int c  = (p & 31) * 4;            // col offset 0..124
            pb[2 * i]     = *(const float4*)(W + (size_t)(k0 + 2 * pr)     * N + col0 + c);
            pb[2 * i + 1] = *(const float4*)(W + (size_t)(k0 + 2 * pr + 1) * N + col0 + c);
        }
    };

    auto lds_store = [&](int buf) {
        if (STAGE == 1) {
            #pragma unroll
            for (int i = 0; i < 4; ++i) {
                int f4 = i * 256 + tid;
                int r  = f4 >> 3;
                int c  = (f4 & 7) * 4;
                u32x2 p;
                p.x = pack2bf(pa[i].x, pa[i].y);
                p.y = pack2bf(pa[i].z, pa[i].w);
                *(u32x2*)&ldsA[buf][r * BK + c] = p;
            }
        } else {
            #pragma unroll
            for (int i = 0; i < 2; ++i) {
                int ch = i * 256 + tid;
                int r  = ch >> 2;
                int c  = (ch & 3) * 8;
                *(u32x4*)&ldsA[buf][r * BK + c] = pa2[i];
            }
        }
        // transpose W patch: K-pair for column (c+j) is contiguous in LDS
        #pragma unroll
        for (int i = 0; i < 2; ++i) {
            int p  = i * 256 + tid;
            int pr = p >> 5;
            int c  = (p & 31) * 4;
            const float* lo = (const float*)&pb[2 * i];
            const float* hi = (const float*)&pb[2 * i + 1];
            #pragma unroll
            for (int j = 0; j < 4; ++j) {
                *(unsigned int*)&ldsB[buf][(c + j) * BK + 2 * pr] =
                    pack2bf(lo[j], hi[j]);
            }
        }
    };

    // ---- pipeline prologue ----
    global_load(0);
    lds_store(0);
    __syncthreads();

    const int niter = K / BK;
    for (int it = 0; it < niter; ++it) {
        const int cur = it & 1;

        // Issue next K-tile's global loads early: latency hides under WMMAs.
        if (it + 1 < niter) global_load((it + 1) * BK);

        // Prime GL2 for the weight panel two K-steps ahead (global_prefetch_b8).
        if (it + 2 < niter) {
            const char* pf = (const char*)(W + (size_t)((it + 2) * BK + (tid >> 3)) * N
                                             + col0 + (tid & 7) * 16);
            __builtin_prefetch(pf, 0, 1);
        }

        // ---- fragments + 8x v_wmma_f32_16x16x32_bf16 from buf[cur] ----
        v16bf afrag[2], bfrag[4];
        #pragma unroll
        for (int t = 0; t < 2; ++t)
            afrag[t] = load_frag(&ldsA[cur][0], wm * 32 + t * 16, lane);
        #pragma unroll
        for (int n = 0; n < 4; ++n)
            bfrag[n] = load_frag(&ldsB[cur][0], wn * 64 + n * 16, lane);

        #pragma unroll
        for (int t = 0; t < 2; ++t)
            #pragma unroll
            for (int n = 0; n < 4; ++n)
                acc[t][n] = __builtin_amdgcn_wmma_f32_16x16x32_bf16(
                    /*neg_a=*/false, afrag[t],
                    /*neg_b=*/false, bfrag[n],
                    /*c_mod=*/(short)0, acc[t][n],
                    /*reuse_a=*/false, /*reuse_b=*/false);

        // Write next tile into the other buffer; single barrier per K-step
        // (reads of buf b in iter i and writes of buf b in iter i+1 are
        //  separated by exactly this barrier).
        if (it + 1 < niter) lds_store(cur ^ 1);
        __syncthreads();
    }

    // ---- epilogue: C/D layout = VGPR r -> row r + 8*(lane>=16); lane%16 -> col
    const int lrow = (lane >> 4) * 8;
    const int lcol = lane & 15;
    #pragma unroll
    for (int t = 0; t < 2; ++t) {
        const int rbase = row0 + wm * 32 + t * 16 + lrow;
        #pragma unroll
        for (int n = 0; n < 4; ++n) {
            const int col = col0 + wn * 64 + n * 16 + lcol;
            const float bs = bias[col];
            #pragma unroll
            for (int r = 0; r < 8; ++r) {
                float v = acc[t][n][r];
                if (STAGE == 1) {
                    float g = gelu_tanh(v + bs);
                    ((unsigned short*)outp)[(size_t)(rbase + r) * N + col] = f2bf(g);
                } else {
                    // nan_to_num(nan=0, +inf=0, -inf=0), then + b2
                    unsigned int u = __float_as_uint(v);
                    if ((u & 0x7F800000u) == 0x7F800000u) v = 0.0f;
                    ((float*)outp)[(size_t)(rbase + r) * N + col] = v + bs;
                }
            }
        }
    }
}

extern "C" void kernel_launch(void* const* d_in, const int* in_sizes, int n_in,
                              void* d_out, int out_size, void* d_ws, size_t ws_size,
                              hipStream_t stream) {
    const float* x  = (const float*)d_in[0];   // [T, D]
    const float* W1 = (const float*)d_in[1];   // [D, DFF]
    const float* b1 = (const float*)d_in[2];   // [DFF]
    const float* W2 = (const float*)d_in[3];   // [DFF, D]
    const float* b2 = (const float*)d_in[4];   // [D]
    float* y = (float*)d_out;                  // [T, D]

    // intermediate h in bf16: T * DFF * 2 = 134 MB in workspace
    unsigned short* h = (unsigned short*)d_ws;

    dim3 blk(256);

    // Stage 1: h = gelu(x @ W1 + b1)   M=T, N=DFF, K=DMODEL
    dim3 g1(T_TOK / BM, DFF / BN);      // (32, 128), M-major for L2 reuse
    ffn_gemm_kernel<1><<<g1, blk, 0, stream>>>(x, W1, b1, h,
                                               T_TOK, DFF, DMODEL);

    // Stage 2: y = nan_guard(h @ W2) + b2   M=T, N=DMODEL, K=DFF
    dim3 g2(T_TOK / BM, DMODEL / BN);   // (32, 32)
    ffn_gemm_kernel<2><<<g2, blk, 0, stream>>>(h, W2, b2, y,
                                               T_TOK, DMODEL, DFF);
}
